// FrameTaggingTransformerWithCRF_14370960572741
// MI455X (gfx1250) — matrix operand order repaired and checked by reading, not compiled, not run on hardware
//
#include <hip/hip_runtime.h>
#include <math.h>

// ---------------------------------------------------------------------------
// MI455X (gfx1250): all large GEMMs + attention on v_wmma_f32_16x16x32_bf16
// (f32 accumulate). Weight tiles staged into LDS by the Tensor Data Mover
// (tensor_load_to_lds + s_wait_tensorcnt), double-buffered, one DMA in flight.
// B=64 S=512 D=512 H=8 dh=64 FF=2048 L=4 T=32, M = B*S = 32768.
// ---------------------------------------------------------------------------

typedef __attribute__((ext_vector_type(16))) __bf16 v16bf;
typedef __attribute__((ext_vector_type(8)))  float  v8f;
typedef unsigned int v4u __attribute__((ext_vector_type(4)));
typedef int          v8i __attribute__((ext_vector_type(8)));
typedef int          v4i __attribute__((ext_vector_type(4)));

union FragB16 {
    v16bf v;
    uint4 q[2];
    unsigned short s[16];
};

__device__ __forceinline__ unsigned short f2bf(float f) {
    unsigned u = __float_as_uint(f);
    u += 0x7FFFu + ((u >> 16) & 1u);   // round-to-nearest-even
    return (unsigned short)(u >> 16);
}

#define WMMA_BF16(A, B, C) \
    __builtin_amdgcn_wmma_f32_16x16x32_bf16(false, (A), false, (B), (short)0, (C), false, false)

#if defined(__has_builtin)
#if __has_builtin(__builtin_amdgcn_tensor_load_to_lds)
#define HAVE_TDM 1
#endif
#endif

// Issue a 2D TDM load: tile_rows x tile_cols bf16 elements, global row stride
// row_stride elems, into contiguous LDS at lds_off. D# per CDNA5 ISA §8.3/8.4.
__device__ __forceinline__ void tdm_load_2d(unsigned lds_off, const void* gptr,
                                            unsigned tile_rows, unsigned tile_cols,
                                            unsigned row_stride) {
#ifdef HAVE_TDM
    const unsigned long long ga = (unsigned long long)(uintptr_t)gptr;
    v4u g0;
    g0[0] = 1u;                                   // count=1, user descriptor
    g0[1] = lds_off;                              // lds_addr (bytes)
    g0[2] = (unsigned)(ga & 0xFFFFFFFFu);         // global_addr[31:0]
    g0[3] = (unsigned)((ga >> 32) & 0x1FFFFFFu)   // global_addr[56:32]
          | (2u << 30);                           // type = 2 ("image")
    v8i g1;
    g1[0] = (int)(1u << 16);                      // data_size=1 (2 bytes)
    g1[1] = (int)((tile_cols & 0xFFFFu) << 16);   // tensor_dim0[15:0] @ bits63:48
    g1[2] = (int)(((tile_cols >> 16) & 0xFFFFu)   // tensor_dim0[31:16]
          | ((tile_rows & 0xFFFFu) << 16));       // tensor_dim1[15:0]
    g1[3] = (int)(((tile_rows >> 16) & 0xFFFFu)   // tensor_dim1[31:16]
          | ((tile_cols & 0xFFFFu) << 16));       // tile_dim0 @ bits127:112
    g1[4] = (int)(tile_rows & 0xFFFFu);           // tile_dim1 (tile_dim2=0)
    g1[5] = (int)row_stride;                      // tensor_dim0_stride[31:0]
    g1[6] = 0;                                    // stride[47:32]=0, d1stride lo=0
    g1[7] = 0;
    v4i gz = {0, 0, 0, 0};
#if __clang_major__ >= 23
    v8i gz8 = {0, 0, 0, 0, 0, 0, 0, 0};
    __builtin_amdgcn_tensor_load_to_lds(g0, g1, gz, gz, gz8, 0);
#else
    __builtin_amdgcn_tensor_load_to_lds(g0, g1, gz, gz, 0);
#endif
#else
    (void)lds_off; (void)gptr; (void)tile_rows; (void)tile_cols; (void)row_stride;
#endif
}

// ---------------------------------------------------------------------------
// f32 -> bf16 weight conversion
// ---------------------------------------------------------------------------
__global__ void k_f32_to_bf16(const float* __restrict__ in,
                              unsigned short* __restrict__ out, size_t n) {
    size_t i = (size_t)blockIdx.x * blockDim.x + threadIdx.x;
    if (i < n) out[i] = f2bf(in[i]);
}

// ---------------------------------------------------------------------------
// Feature assembly + input projection + sqrt(D) scale + positional encoding
// ---------------------------------------------------------------------------
__global__ void k_embed(const float* __restrict__ ssim,      // [B,S,6]
                        const float* __restrict__ flags,     // [B,S,1]
                        const float* __restrict__ emb,       // [NOP,8]
                        const float* __restrict__ Wp,        // [512,15]
                        const float* __restrict__ bp,        // [512]
                        const int*   __restrict__ op_types,  // [B]
                        float* __restrict__ outF,            // [M,512]
                        unsigned short* __restrict__ outB) { // [M,512]
    __shared__ float feat[16];
    const int bs = blockIdx.x;
    const int b  = bs >> 9;
    const int s  = bs & 511;
    if (threadIdx.x < 6)        feat[threadIdx.x] = ssim[(size_t)bs * 6 + threadIdx.x];
    else if (threadIdx.x < 14)  feat[threadIdx.x] = emb[op_types[b] * 8 + threadIdx.x - 6];
    else if (threadIdx.x == 14) feat[14] = flags[bs];
    __syncthreads();
    for (int d = threadIdx.x; d < 512; d += blockDim.x) {
        float acc = bp[d];
        const float* w = Wp + d * 15;
        #pragma unroll
        for (int j = 0; j < 15; ++j) acc += w[j] * feat[j];
        const int   i2  = (d >> 1) << 1;
        const float div = expf(-logf(10000.f) * (float)i2 * (1.f / 512.f));
        const float ang = (float)s * div;
        const float pe  = (d & 1) ? cosf(ang) : sinf(ang);
        const float v   = acc * 22.62741699796952f + pe;   // sqrt(512)
        outF[(size_t)bs * 512 + d] = v;
        outB[(size_t)bs * 512 + d] = f2bf(v);
    }
}

// ---------------------------------------------------------------------------
// WMMA GEMM, NT in {2,4} 16-wide N tiles per wave:
//   Y[M,N] = A[M,K] @ W[N,K]^T + bias
// Block = 256 threads = 8 waves; wave w owns rows [m0+16w, m0+16w+16).
// Weight tile (NT*16 rows x 32 k) staged to LDS by TDM from wave 0,
// double-buffered; one tensor DMA in flight overlapping WMMA compute.
// A-fragments software-pipelined via a single loop-carried pointer.
// Epilogue: 0 outF=y | 1 outB=bf16(y) | 2 outB=bf16(gelu(y)) | 3 outF=y+resid
// ---------------------------------------------------------------------------
template <int NT>
__global__ __launch_bounds__(256, 1)
void k_gemm_bf16(const unsigned short* __restrict__ A,   // [M,K] bf16
                 const unsigned short* __restrict__ W,   // [N,K] bf16
                 const float* __restrict__ bias,         // [N]
                 const float* __restrict__ resid,        // [M,N] or null
                 float* __restrict__ outF,
                 unsigned short* __restrict__ outB,
                 int M, int N, int K, int epi) {
    __shared__ __align__(16) unsigned short wtile[2][NT * 16 * 32];
    const int lane = threadIdx.x & 31;
    const int wave = threadIdx.x >> 5;
    const int mrow = lane & 15;
    const int kh   = lane >> 4;
    const int m0   = blockIdx.x * 128 + wave * 16;
    const int n0   = blockIdx.y * (NT * 16);

    v8f acc[NT];
    #pragma unroll
    for (int t = 0; t < NT; ++t)
        #pragma unroll
        for (int r = 0; r < 8; ++r) acc[t][r] = 0.f;

    const unsigned short* wbase = W + (size_t)n0 * K;
    const unsigned lds0 = (unsigned)(uintptr_t)(void*)&wtile[0][0];
    const unsigned lds1 = (unsigned)(uintptr_t)(void*)&wtile[1][0];

    // lane-fixed bases; only `ap` is loop-carried (+32 elems / iter)
    const unsigned short* ap  = A + (size_t)(m0 + mrow) * K + kh * 16;
    const unsigned short* tb0 = &wtile[0][mrow * 32 + kh * 16];
    const unsigned short* tb1 = &wtile[1][mrow * 32 + kh * 16];

    // prologue: issue first weight tile, preload first A fragment
    if (wave == 0) tdm_load_2d(lds0, wbase, NT * 16, 32, (unsigned)K);
#ifndef HAVE_TDM
    for (int c = threadIdx.x; c < NT * 16 * 32 / 8; c += 256)
        *(uint4*)&wtile[0][c * 8] =
            *(const uint4*)&wbase[(c / 4) * K + (c & 3) * 8];
#endif
    FragB16 fa;
    fa.q[0] = *reinterpret_cast<const uint4*>(ap);
    fa.q[1] = *reinterpret_cast<const uint4*>(ap + 8);

    int buf = 0;
    #pragma unroll 1
    for (int kb = 0; kb < K; kb += 32) {
        if (wave == 0) __builtin_amdgcn_s_wait_tensorcnt(0);
        __syncthreads();                      // tile[buf] ready; tile[buf^1] free
        const int more = (kb + 32 < K);
        if (more && wave == 0)
            tdm_load_2d(buf ? lds0 : lds1, wbase + (size_t)(kb + 32),
                        NT * 16, 32, (unsigned)K);
#ifndef HAVE_TDM
        if (more)
            for (int c = threadIdx.x; c < NT * 16 * 32 / 8; c += 256)
                *(uint4*)&wtile[buf ^ 1][c * 8] =
                    *(const uint4*)&wbase[(c / 4) * K + kb + 32 + (c & 3) * 8];
#endif
        // prefetch next A fragment (clamped pointer on last iteration)
        const unsigned short* apn = more ? ap + 32 : ap;
        FragB16 fa_nxt;
        fa_nxt.q[0] = *reinterpret_cast<const uint4*>(apn);
        fa_nxt.q[1] = *reinterpret_cast<const uint4*>(apn + 8);
        // branch-free WMMA over NT tiles from LDS (immediate DS offsets)
        const unsigned short* tb = buf ? tb1 : tb0;
        #pragma unroll
        for (int t = 0; t < NT; ++t) {
            FragB16 fb;
            fb.q[0] = *reinterpret_cast<const uint4*>(tb + t * 512);
            fb.q[1] = *reinterpret_cast<const uint4*>(tb + t * 512 + 8);
            acc[t] = WMMA_BF16(fa.v, fb.v, acc[t]);
        }
        fa  = fa_nxt;
        ap  = apn;
        buf ^= 1;
    }

    #pragma unroll
    for (int t = 0; t < NT; ++t) {
        const int col = n0 + t * 16 + mrow;
        const float bv = bias ? bias[col] : 0.f;
        #pragma unroll
        for (int r = 0; r < 8; ++r) {
            const int row = m0 + r + 8 * kh;
            float v = acc[t][r] + bv;
            const size_t idx = (size_t)row * N + col;
            if (epi == 3)      { outF[idx] = v + resid[idx]; }
            else if (epi == 2) { v = 0.5f * v * (1.f + erff(v * 0.7071067811865475f));
                                 outB[idx] = f2bf(v); }
            else if (epi == 1) { outB[idx] = f2bf(v); }
            else               { outF[idx] = v; }
        }
    }
}

// ---------------------------------------------------------------------------
// V transpose: qkv[b,s, 2D + h*64 + d] -> vt[(b*H+h)*64 + d, s]  (bf16 bits)
// ---------------------------------------------------------------------------
__global__ void k_vtrans(const unsigned short* __restrict__ qkv,
                         unsigned short* __restrict__ vt) {
    const size_t i = (size_t)blockIdx.x * blockDim.x + threadIdx.x;
    if (i >= (size_t)32768 * 512) return;
    const int    d  = (int)(i & 511);
    const size_t bs = i >> 9;
    const int    s  = (int)(bs & 511);
    const int    b  = (int)(bs >> 9);
    const int    h  = d >> 6;
    const int    dd = d & 63;
    vt[(((size_t)(b * 8 + h) * 64) + dd) * 512 + s] = qkv[bs * 1536 + 1024 + d];
}

// ---------------------------------------------------------------------------
// Flash attention, one (b,h) per blockIdx.x, 128 query rows per block,
// 16 query rows per wave, key blocks of 32, WMMA for QK^T and P@V.
// Loop-carried pointers (pk, pv); all tile offsets are immediates.
// P transposed C-layout -> A-layout through a wave-private LDS tile.
// ---------------------------------------------------------------------------
__global__ __launch_bounds__(256, 1)
void k_attn(const unsigned short* __restrict__ qkv,  // [M,1536]
            const unsigned short* __restrict__ vt,   // [B*H,64,512]
            unsigned short* __restrict__ ctx) {      // [M,512]
    __shared__ __align__(16) unsigned short pshm[8][16 * 32];
    const int lane = threadIdx.x & 31;
    const int wave = threadIdx.x >> 5;
    const int mrow = lane & 15;
    const int kh   = lane >> 4;
    const int bh   = blockIdx.x;           // b*8 + h
    const int b    = bh >> 3;
    const int h    = bh & 7;
    const int q0   = blockIdx.y * 128 + wave * 16;

    FragB16 fq0, fq1;
    {
        const unsigned short* qb =
            qkv + ((size_t)b * 512 + q0 + mrow) * 1536 + h * 64 + kh * 16;
        fq0.q[0] = *reinterpret_cast<const uint4*>(qb);
        fq0.q[1] = *reinterpret_cast<const uint4*>(qb + 8);
        fq1.q[0] = *reinterpret_cast<const uint4*>(qb + 32);
        fq1.q[1] = *reinterpret_cast<const uint4*>(qb + 40);
    }

    float mrun[8], lrun[8];
    v8f o[4];
    #pragma unroll
    for (int r = 0; r < 8; ++r) { mrun[r] = -1e30f; lrun[r] = 0.f; }
    #pragma unroll
    for (int t = 0; t < 4; ++t)
        #pragma unroll
        for (int r = 0; r < 8; ++r) o[t][r] = 0.f;

    // loop-carried pointers; everything else is an immediate offset
    const unsigned short* pk =
        qkv + ((size_t)b * 512 + mrow) * 1536 + 512 + h * 64 + kh * 16;
    const unsigned short* pv =
        vt + (size_t)bh * 64 * 512 + (size_t)mrow * 512 + kh * 16;
    unsigned short* pw = pshm[wave] + (8 * kh) * 32 + mrow;
    const unsigned short* pr = pshm[wave] + mrow * 32 + kh * 16;

    #pragma unroll 1
    for (int kb = 0; kb < 512; kb += 32) {
        v8f sa[2];
        #pragma unroll
        for (int tk = 0; tk < 2; ++tk) {
            #pragma unroll
            for (int r = 0; r < 8; ++r) sa[tk][r] = 0.f;
            const unsigned short* kp = pk + tk * (16 * 1536);
            FragB16 fk, fk2;
            fk.q[0]  = *reinterpret_cast<const uint4*>(kp);
            fk.q[1]  = *reinterpret_cast<const uint4*>(kp + 8);
            fk2.q[0] = *reinterpret_cast<const uint4*>(kp + 32);
            fk2.q[1] = *reinterpret_cast<const uint4*>(kp + 40);
            sa[tk] = WMMA_BF16(fq0.v, fk.v, sa[tk]);
            sa[tk] = WMMA_BF16(fq1.v, fk2.v, sa[tk]);
        }
        #pragma unroll
        for (int r = 0; r < 8; ++r) {
            float s0 = sa[0][r] * 0.125f;          // 1/sqrt(64)
            float s1 = sa[1][r] * 0.125f;
            float mx = fmaxf(s0, s1);
            #pragma unroll
            for (int off = 1; off < 16; off <<= 1)
                mx = fmaxf(mx, __shfl_xor(mx, off, 32));
            const float nm = fmaxf(mrun[r], mx);
            const float p0 = __expf(s0 - nm);
            const float p1 = __expf(s1 - nm);
            float rs = p0 + p1;
            #pragma unroll
            for (int off = 1; off < 16; off <<= 1)
                rs += __shfl_xor(rs, off, 32);
            const float corr = __expf(mrun[r] - nm);
            lrun[r] = lrun[r] * corr + rs;
            mrun[r] = nm;
            #pragma unroll
            for (int t = 0; t < 4; ++t) o[t][r] *= corr;
            pw[r * 32]      = f2bf(p0);
            pw[r * 32 + 16] = f2bf(p1);
        }
        asm volatile("s_wait_dscnt 0x0" ::: "memory");   // LDS store->load, same wave
        FragB16 fp;
        fp.q[0] = *reinterpret_cast<const uint4*>(pr);
        fp.q[1] = *reinterpret_cast<const uint4*>(pr + 8);
        #pragma unroll
        for (int t = 0; t < 4; ++t) {
            FragB16 fv;
            fv.q[0] = *reinterpret_cast<const uint4*>(pv + t * (16 * 512));
            fv.q[1] = *reinterpret_cast<const uint4*>(pv + t * (16 * 512) + 8);
            o[t] = WMMA_BF16(fp.v, fv.v, o[t]);
        }
        pk += 32 * 1536;
        pv += 32;
    }

    #pragma unroll
    for (int t = 0; t < 4; ++t)
        #pragma unroll
        for (int r = 0; r < 8; ++r) {
            const int row = q0 + r + 8 * kh;
            const float val = o[t][r] / lrun[r];
            ctx[((size_t)b * 512 + row) * 512 + h * 64 + t * 16 + mrow] = f2bf(val);
        }
}

// ---------------------------------------------------------------------------
// LayerNorm over D=512, one block per row; writes f32 residual stream + bf16.
// ---------------------------------------------------------------------------
__global__ void k_layernorm(const float* __restrict__ in,
                            const float* __restrict__ g,
                            const float* __restrict__ be,
                            float* __restrict__ outF,
                            unsigned short* __restrict__ outB) {
    __shared__ float s_sum[256];
    __shared__ float s_sq[256];
    const int row = blockIdx.x;
    const float* x = in + (size_t)row * 512;
    float sum = 0.f, sq = 0.f;
    for (int d = threadIdx.x; d < 512; d += blockDim.x) {
        const float v = x[d];
        sum += v; sq += v * v;
    }
    s_sum[threadIdx.x] = sum; s_sq[threadIdx.x] = sq;
    __syncthreads();
    for (int s = blockDim.x >> 1; s > 0; s >>= 1) {
        if (threadIdx.x < (unsigned)s) {
            s_sum[threadIdx.x] += s_sum[threadIdx.x + s];
            s_sq[threadIdx.x]  += s_sq[threadIdx.x + s];
        }
        __syncthreads();
    }
    const float mu   = s_sum[0] * (1.f / 512.f);
    const float var  = s_sq[0] * (1.f / 512.f) - mu * mu;
    const float rstd = rsqrtf(var + 1e-5f);
    for (int d = threadIdx.x; d < 512; d += blockDim.x) {
        const float v = (x[d] - mu) * rstd * g[d] + be[d];
        outF[(size_t)row * 512 + d] = v;
        outB[(size_t)row * 512 + d] = f2bf(v);
    }
}

// ---------------------------------------------------------------------------
// CRF NLL: one wave32 per batch, lane j = tag j (T == 32 == wave width).
// ---------------------------------------------------------------------------
__global__ void k_crf(const float* __restrict__ emis,     // [B,S,32]
                      const float* __restrict__ start,    // [32]
                      const float* __restrict__ endv,     // [32]
                      const float* __restrict__ trans,    // [32,32]
                      const int*   __restrict__ labels,   // [B,S]
                      float* __restrict__ loss) {         // [B]
    __shared__ float T_[32 * 33];
    const int b = blockIdx.x;
    const int j = threadIdx.x;
    for (int i = 0; i < 32; ++i) T_[i * 33 + j] = trans[i * 32 + j];
    __syncthreads();
    const float* eb = emis + (size_t)b * 512 * 32;
    const int*   lb = labels + b * 512;

    float gpart = 0.f;
    for (int t = j; t < 512; t += 32) {
        gpart += eb[(size_t)t * 32 + lb[t]];
        if (t >= 1) gpart += trans[lb[t - 1] * 32 + lb[t]];
    }
    #pragma unroll
    for (int off = 16; off; off >>= 1) gpart += __shfl_xor(gpart, off, 32);
    const float numer = gpart + start[lb[0]] + endv[lb[511]];

    float alpha = start[j] + eb[j];
    #pragma unroll 1
    for (int t = 1; t < 512; ++t) {
        float m = -3.4e38f;
        #pragma unroll 4
        for (int i = 0; i < 32; ++i) {
            const float ai = __shfl(alpha, i, 32);
            m = fmaxf(m, ai + T_[i * 33 + j]);
        }
        float ssum = 0.f;
        #pragma unroll 4
        for (int i = 0; i < 32; ++i) {
            const float ai = __shfl(alpha, i, 32);
            ssum += __expf(ai + T_[i * 33 + j] - m);
        }
        alpha = m + __logf(ssum) + eb[(size_t)t * 32 + j];
    }
    float v = alpha + endv[j];
    float m2 = v;
    #pragma unroll
    for (int off = 16; off; off >>= 1) m2 = fmaxf(m2, __shfl_xor(m2, off, 32));
    float s2 = __expf(v - m2);
    #pragma unroll
    for (int off = 16; off; off >>= 1) s2 += __shfl_xor(s2, off, 32);
    if (j == 0) loss[b] = -(numer - (m2 + __logf(s2)));
}

__global__ void k_final(const float* __restrict__ loss, float* __restrict__ out) {
    __shared__ float sh[64];
    sh[threadIdx.x] = loss[threadIdx.x];
    __syncthreads();
    if (threadIdx.x == 0) {
        float s = 0.f;
        for (int i = 0; i < 64; ++i) s += sh[i];
        out[0] = s * (1.f / 64.f);
    }
}

// ---------------------------------------------------------------------------
// Host orchestration
// ---------------------------------------------------------------------------
extern "C" void kernel_launch(void* const* d_in, const int* in_sizes, int n_in,
                              void* d_out, int out_size, void* d_ws, size_t ws_size,
                              hipStream_t stream) {
    (void)in_sizes; (void)n_in; (void)out_size; (void)ws_size;
    const float* ssim      = (const float*)d_in[0];
    const float* flags     = (const float*)d_in[1];
    const float* emb       = (const float*)d_in[2];
    const float* Wp        = (const float*)d_in[3];
    const float* bp        = (const float*)d_in[4];
    const float* Wqkv      = (const float*)d_in[5];
    const float* bqkv      = (const float*)d_in[6];
    const float* Wo        = (const float*)d_in[7];
    const float* bo        = (const float*)d_in[8];
    const float* W1        = (const float*)d_in[9];
    const float* b1        = (const float*)d_in[10];
    const float* W2        = (const float*)d_in[11];
    const float* b2        = (const float*)d_in[12];
    const float* g1        = (const float*)d_in[13];
    const float* be1       = (const float*)d_in[14];
    const float* g2        = (const float*)d_in[15];
    const float* be2       = (const float*)d_in[16];
    const float* Wtag      = (const float*)d_in[17];
    const float* btag      = (const float*)d_in[18];
    const float* crf_start = (const float*)d_in[19];
    const float* crf_end   = (const float*)d_in[20];
    const float* crf_trans = (const float*)d_in[21];
    const int*   op_types  = (const int*)d_in[22];
    const int*   labels    = (const int*)d_in[23];
    float* out = (float*)d_out;

    const int M = 64 * 512;   // 32768

    char* base = (char*)d_ws;
    auto alloc = [&](size_t bytes) -> char* {
        char* p = base;
        base += (bytes + 255) & ~(size_t)255;
        return p;
    };
    float*          xf     = (float*)alloc((size_t)M * 512 * 4);
    float*          tmpf   = (float*)alloc((size_t)M * 512 * 4);
    unsigned short* xb     = (unsigned short*)alloc((size_t)M * 512 * 2);
    unsigned short* big    = (unsigned short*)alloc((size_t)M * 2048 * 2);
    unsigned short* ctxb   = (unsigned short*)alloc((size_t)M * 512 * 2);
    unsigned short* vtb    = (unsigned short*)alloc((size_t)M * 512 * 2);
    float*          emis   = (float*)alloc((size_t)M * 32 * 4);
    unsigned short* Wqkv_b = (unsigned short*)alloc((size_t)4 * 1536 * 512 * 2);
    unsigned short* Wo_b   = (unsigned short*)alloc((size_t)4 * 512 * 512 * 2);
    unsigned short* W1_b   = (unsigned short*)alloc((size_t)4 * 2048 * 512 * 2);
    unsigned short* W2_b   = (unsigned short*)alloc((size_t)4 * 512 * 2048 * 2);
    unsigned short* Wtag_b = (unsigned short*)alloc((size_t)32 * 512 * 2);
    float*          lossb  = (float*)alloc(64 * 4);

    auto cvt = [&](const float* src, unsigned short* dst, size_t n) {
        k_f32_to_bf16<<<(unsigned)((n + 255) / 256), 256, 0, stream>>>(src, dst, n);
    };
    cvt(Wqkv, Wqkv_b, (size_t)4 * 1536 * 512);
    cvt(Wo,   Wo_b,   (size_t)4 * 512 * 512);
    cvt(W1,   W1_b,   (size_t)4 * 2048 * 512);
    cvt(W2,   W2_b,   (size_t)4 * 512 * 2048);
    cvt(Wtag, Wtag_b, (size_t)32 * 512);

    k_embed<<<M, 128, 0, stream>>>(ssim, flags, emb, Wp, bp, op_types, xf, xb);

    for (int l = 0; l < 4; ++l) {
        // QKV projection -> big[M,1536] bf16
        k_gemm_bf16<4><<<dim3(M / 128, 1536 / 64), 256, 0, stream>>>(
            xb, Wqkv_b + (size_t)l * 1536 * 512, bqkv + l * 1536,
            nullptr, nullptr, big, M, 1536, 512, 1);
        // V transpose for contiguous B-frags in P@V
        k_vtrans<<<(M * 512) / 256, 256, 0, stream>>>(big, vtb);
        // flash attention -> ctxb[M,512] bf16
        k_attn<<<dim3(64 * 8, 512 / 128), 256, 0, stream>>>(big, vtb, ctxb);
        // output projection + residual -> tmpf f32
        k_gemm_bf16<4><<<dim3(M / 128, 512 / 64), 256, 0, stream>>>(
            ctxb, Wo_b + (size_t)l * 512 * 512, bo + l * 512,
            xf, tmpf, nullptr, M, 512, 512, 3);
        k_layernorm<<<M, 256, 0, stream>>>(tmpf, g1 + l * 512, be1 + l * 512, xf, xb);
        // FFN up + exact GELU -> big[M,2048] bf16
        k_gemm_bf16<4><<<dim3(M / 128, 2048 / 64), 256, 0, stream>>>(
            xb, W1_b + (size_t)l * 2048 * 512, b1 + l * 2048,
            nullptr, nullptr, big, M, 2048, 512, 2);
        // FFN down + residual -> tmpf f32
        k_gemm_bf16<4><<<dim3(M / 128, 512 / 64), 256, 0, stream>>>(
            big, W2_b + (size_t)l * 512 * 2048, b2 + l * 512,
            xf, tmpf, nullptr, M, 512, 2048, 3);
        k_layernorm<<<M, 256, 0, stream>>>(tmpf, g2 + l * 512, be2 + l * 512, xf, xb);
    }

    // emissions [M,32] f32: NT=2 (N = 32 exactly, branch-free)
    k_gemm_bf16<2><<<dim3(M / 128, 1), 256, 0, stream>>>(
        xb, Wtag_b, btag, nullptr, emis, nullptr, M, 32, 512, 0);

    k_crf<<<64, 32, 0, stream>>>(emis, crf_start, crf_end, crf_trans, labels, lossb);
    k_final<<<1, 64, 0, stream>>>(lossb, out);
}
